// RGIN_Net_10634339025230
// MI455X (gfx1250) — matrix operand order repaired and checked by reading, not compiled, not run on hardware
//
#include <hip/hip_runtime.h>
#include <hip/hip_bf16.h>

#define NN 100000
#define EE 1250000
#define RR 8
#define DD 64
#define LL 2

typedef __attribute__((ext_vector_type(16))) _Float16 v16h;
typedef __attribute__((ext_vector_type(8)))  _Float16 v8h;
typedef __attribute__((ext_vector_type(8)))  float    v8f;

__device__ inline v8f wmma_f16(v16h a, v16h b, v8f c) {
    // D = A(16x32 f16) * B(32x16 f16) + C(16x16 f32)
    return __builtin_amdgcn_wmma_f32_16x16x32_f16(
        /*neg_a=*/false, a, /*neg_b=*/false, b,
        /*c_mod=*/(short)0, c, /*reuse_a=*/false, /*reuse_b=*/false);
}

__device__ inline void pack8(v16h& a, int base, float4 lo, float4 hi) {
    a[base + 0] = (_Float16)lo.x; a[base + 1] = (_Float16)lo.y;
    a[base + 2] = (_Float16)lo.z; a[base + 3] = (_Float16)lo.w;
    a[base + 4] = (_Float16)hi.x; a[base + 5] = (_Float16)hi.y;
    a[base + 6] = (_Float16)hi.z; a[base + 7] = (_Float16)hi.w;
}

// ---------------------------------------------------------------------------
// K_w: convert weights to f16, transposed to [out][in] for contiguous
// B-fragment loads. 0..15 = W1[l][r], 16..31 = W2[l][r], 32..33 = rW1[l],
// 34..35 = rW2[l]
// ---------------------------------------------------------------------------
__global__ void k_weights(const float* __restrict__ W1, const float* __restrict__ W2,
                          const float* __restrict__ rW1, const float* __restrict__ rW2,
                          _Float16* __restrict__ W1t, _Float16* __restrict__ W2t,
                          _Float16* __restrict__ rW1t, _Float16* __restrict__ rW2t) {
    int mat = blockIdx.x;
    const float* src; _Float16* dst;
    if (mat < 16)      { src = W1  + (size_t)mat        * 4096; dst = W1t  + (size_t)mat        * 4096; }
    else if (mat < 32) { src = W2  + (size_t)(mat - 16) * 4096; dst = W2t  + (size_t)(mat - 16) * 4096; }
    else if (mat < 34) { src = rW1 + (size_t)(mat - 32) * 4096; dst = rW1t + (size_t)(mat - 32) * 4096; }
    else               { src = rW2 + (size_t)(mat - 34) * 4096; dst = rW2t + (size_t)(mat - 34) * 4096; }
    for (int e = threadIdx.x; e < 4096; e += blockDim.x) {
        int n = e >> 6, k = e & 63;
        dst[e] = (_Float16)src[k * 64 + n];   // dst[n][k] = src[k][n]
    }
}

// ---------------------------------------------------------------------------
// K1: edge scatter-add  h[type][dst][:] += x[src][:]; 16 threads/edge.
// ---------------------------------------------------------------------------
__global__ void k_scatter(const float* __restrict__ x, const int* __restrict__ ei,
                          const int* __restrict__ et, float* __restrict__ h) {
    long long tid = (long long)blockIdx.x * blockDim.x + threadIdx.x;
    long long e = tid >> 4;
    int part = (int)(tid & 15);
    if (e >= EE) return;
    int src = ei[e];
    int dst = ei[EE + e];
    int t   = et[e];
    const float* xp = x + (size_t)src * DD + part * 4;
    float* hp = h + ((size_t)t * NN + dst) * DD + part * 4;
    atomicAdd(hp + 0, xp[0]);
    atomicAdd(hp + 1, xp[1]);
    atomicAdd(hp + 2, xp[2]);
    atomicAdd(hp + 3, xp[3]);
}

// ---------------------------------------------------------------------------
// K2: z = A @ W1 + b1 per branch (blockIdx.y = r; r==8 = root on x).
// One wave: 16x64 tile, 8 WMMAs. Uniform (scalar-branch) fast/slow epilogue,
// LDS transpose for coalesced f16 stores, LDS-reduced BN stats.
// ---------------------------------------------------------------------------
__global__ void k_gemm1(const float* __restrict__ h, const float* __restrict__ xcur,
                        const _Float16* __restrict__ W1t, const _Float16* __restrict__ rW1t,
                        const float* __restrict__ b1, const float* __restrict__ rb1,
                        _Float16* __restrict__ z, float* __restrict__ ssum,
                        float* __restrict__ ssq, int layer) {
    __shared__ _Float16 zstage[8][16][64];   // 16 KB
    __shared__ float    lstat[2][64];        // 512 B

    int r    = blockIdx.y;
    int wave = threadIdx.x >> 5;
    int lane = threadIdx.x & 31;
    int m    = lane & 15;
    int kh   = lane >> 4;
    int nloc = lane & 15;
    int rowBase = blockIdx.x * 128 + wave * 16;
    int row = rowBase + m;
    int rc  = row < NN ? row : NN - 1;
    bool edge = (blockIdx.x == gridDim.x - 1);   // scalar: only last x-block is partial

    if (threadIdx.x < 128) lstat[threadIdx.x >> 6][threadIdx.x & 63] = 0.f;
    __syncthreads();

    const float* A = (r < RR) ? (h + ((size_t)r * NN + rc) * DD)
                              : (xcur + (size_t)rc * DD);
    const float4* A4 = (const float4*)A;
    v16h a0, a1;
    pack8(a0, 0, A4[4 * kh + 0],  A4[4 * kh + 1]);    // K = 16*kh .. +7
    pack8(a0, 8, A4[4 * kh + 2],  A4[4 * kh + 3]);    // K = 16*kh+8 .. +7
    pack8(a1, 0, A4[4 * kh + 8],  A4[4 * kh + 9]);    // K = 32+16*kh .. +7
    pack8(a1, 8, A4[4 * kh + 10], A4[4 * kh + 11]);   // K = 40+16*kh .. +7

    const _Float16* Wt = (r < RR) ? (W1t + ((size_t)layer * RR + r) * 4096)
                                  : (rW1t + (size_t)layer * 4096);
    const float* bv = (r < RR) ? (b1 + (size_t)(layer * RR + r) * DD)
                               : (rb1 + (size_t)layer * DD);

    // all 4 column-tiles of C first, epilogue after (so boundary split is one
    // scalar branch, EXEC stays all-ones through the WMMAs)
    v8f cacc[4];
    #pragma unroll
    for (int nt = 0; nt < 4; nt++) {
        int n = nt * 16 + nloc;
        const _Float16* bp = Wt + (size_t)n * DD + kh * 16;
        v16h bf0 = *(const v16h*)(bp);        // K 0..31 chunk
        v16h bf1 = *(const v16h*)(bp + 32);   // K 32..63 chunk
        v8f c = {};
        c = wmma_f16(a0, bf0, c);
        cacc[nt] = wmma_f16(a1, bf1, c);
    }

    if (!edge) {
        // fast path: every row valid, no masking
        #pragma unroll
        for (int nt = 0; nt < 4; nt++) {
            int n = nt * 16 + nloc;
            float bb = bv[n];
            float s = 0.f, q = 0.f;
            #pragma unroll
            for (int v = 0; v < 8; v++) {
                float val = cacc[nt][v] + bb;
                s += val;
                q += val * val;
                zstage[wave][v + 8 * kh][n] = (_Float16)val;
            }
            atomicAdd(&lstat[0][n], s);
            atomicAdd(&lstat[1][n], q);
        }
    } else {
        #pragma unroll
        for (int nt = 0; nt < 4; nt++) {
            int n = nt * 16 + nloc;
            float bb = bv[n];
            float s = 0.f, q = 0.f;
            #pragma unroll
            for (int v = 0; v < 8; v++) {
                float val = cacc[nt][v] + bb;
                int rm = rowBase + v + 8 * kh;         // C/D: m = vgpr + 8*(lane>>4)
                bool ok = rm < NN;                      // v_cndmask, no branch
                s += ok ? val : 0.f;
                q += ok ? val * val : 0.f;
                zstage[wave][v + 8 * kh][n] = (_Float16)val;
            }
            atomicAdd(&lstat[0][n], s);
            atomicAdd(&lstat[1][n], q);
        }
    }

    // coalesced write-out: 2 lanes per row, 64 B each (4 x b128)
    {
        int rrow = lane >> 1;
        int half = lane & 1;
        int grow = rowBase + rrow;
        const uint4* sp = (const uint4*)&zstage[wave][rrow][half * 32];
        uint4 d0 = sp[0], d1 = sp[1], d2 = sp[2], d3 = sp[3];
        if (grow < NN) {
            uint4* dp = (uint4*)(z + ((size_t)r * NN + grow) * DD + half * 32);
            dp[0] = d0; dp[1] = d1; dp[2] = d2; dp[3] = d3;
        }
    }

    __syncthreads();
    if (threadIdx.x < 64) {
        atomicAdd(&ssum[r * DD + threadIdx.x], lstat[0][threadIdx.x]);
        atomicAdd(&ssq [r * DD + threadIdx.x], lstat[1][threadIdx.x]);
    }
}

// ---------------------------------------------------------------------------
// K3: finalize BN scale/shift per (branch, channel) and fold output biases.
// ---------------------------------------------------------------------------
__global__ void k_stats(const float* __restrict__ ssum, const float* __restrict__ ssq,
                        const float* __restrict__ g1, const float* __restrict__ be1,
                        const float* __restrict__ rg1, const float* __restrict__ rbe1,
                        const float* __restrict__ b2, const float* __restrict__ rb2,
                        const float* __restrict__ bias,
                        float* __restrict__ scale, float* __restrict__ shift,
                        float* __restrict__ csum, int layer) {
    int j = threadIdx.x;
    if (j < (RR + 1) * DD) {
        int r = j >> 6, c = j & 63;
        float mean = ssum[j] * (1.0f / NN);
        float var  = ssq[j]  * (1.0f / NN) - mean * mean;
        float g, be;
        if (r < RR) { g = g1[layer * RR * DD + j];  be = be1[layer * RR * DD + j]; }
        else        { g = rg1[layer * DD + c];      be = rbe1[layer * DD + c]; }
        float s = g * rsqrtf(var + 1e-5f);
        scale[j] = s;
        shift[j] = be - mean * s;
    }
    if (j < DD) {
        float cs = rb2[layer * DD + j] + bias[layer * DD + j];
        #pragma unroll
        for (int r = 0; r < RR; r++) cs += b2[layer * RR * DD + r * DD + j];
        csum[j] = cs;
    }
}

// ---------------------------------------------------------------------------
// K4: out = sum_{rr=0..8} relu(bn(z_rr)) @ W2_rr + csum ; optional final ReLU.
// BN affine + ReLU applied in registers while building f16 A-fragments.
// LDS transpose for coalesced f32 output stores.
// ---------------------------------------------------------------------------
__global__ void k_gemm2(const _Float16* __restrict__ z,
                        const _Float16* __restrict__ W2t, const _Float16* __restrict__ rW2t,
                        const float* __restrict__ scale, const float* __restrict__ shift,
                        const float* __restrict__ csum, float* __restrict__ xnext,
                        int layer, int dorelu) {
    __shared__ float ostage[8][16][64];   // 32 KB

    int wave = threadIdx.x >> 5;
    int lane = threadIdx.x & 31;
    int kh   = lane >> 4;
    int nloc = lane & 15;
    int rowBase = blockIdx.x * 128 + wave * 16;
    int row = rowBase + (lane & 15);
    int rc  = row < NN ? row : NN - 1;

    v8f acc[4] = {};

    for (int rr = 0; rr < RR + 1; rr++) {
        const _Float16* zr = z + ((size_t)rr * NN + rc) * DD;
        const float* sc = scale + rr * DD;
        const float* sh = shift + rr * DD;
        const _Float16* Wt = (rr < RR) ? (W2t + ((size_t)layer * RR + rr) * 4096)
                                       : (rW2t + (size_t)layer * 4096);
        // prefetch next branch's weight tile (global_prefetch_b8)
        if (rr < RR) {
            const _Float16* Wnext = (rr + 1 < RR) ? (W2t + ((size_t)layer * RR + rr + 1) * 4096)
                                                  : (rW2t + (size_t)layer * 4096);
            __builtin_prefetch(Wnext + (size_t)nloc * 4 * DD, 0, 3);
        }
        #pragma unroll
        for (int kc = 0; kc < DD; kc += 32) {
            v8h z0 = *(const v8h*)(zr + kc + 16 * kh);       // K = kc+16kh..+7
            v8h z1 = *(const v8h*)(zr + kc + 16 * kh + 8);   // K = kc+16kh+8..+7
            const float4* sc0 = (const float4*)(sc + kc + 16 * kh);
            const float4* sh0 = (const float4*)(sh + kc + 16 * kh);
            const float4* sc1 = (const float4*)(sc + kc + 16 * kh + 8);
            const float4* sh1 = (const float4*)(sh + kc + 16 * kh + 8);
            float scv0[8], shv0[8], scv1[8], shv1[8];
            *(float4*)&scv0[0] = sc0[0]; *(float4*)&scv0[4] = sc0[1];
            *(float4*)&shv0[0] = sh0[0]; *(float4*)&shv0[4] = sh0[1];
            *(float4*)&scv1[0] = sc1[0]; *(float4*)&scv1[4] = sc1[1];
            *(float4*)&shv1[0] = sh1[0]; *(float4*)&shv1[4] = sh1[1];
            v16h a;
            #pragma unroll
            for (int e = 0; e < 8; e++) {
                float v0 = (float)z0[e] * scv0[e] + shv0[e];
                float v1 = (float)z1[e] * scv1[e] + shv1[e];
                v0 = v0 > 0.f ? v0 : 0.f;
                v1 = v1 > 0.f ? v1 : 0.f;
                a[e]     = (_Float16)v0;
                a[8 + e] = (_Float16)v1;
            }
            #pragma unroll
            for (int nt = 0; nt < 4; nt++) {
                int n = nt * 16 + nloc;
                v16h b = *(const v16h*)(Wt + (size_t)n * DD + kc + kh * 16);
                acc[nt] = wmma_f16(a, b, acc[nt]);
            }
        }
    }

    // epilogue: +csum, optional ReLU, stage to LDS
    #pragma unroll
    for (int nt = 0; nt < 4; nt++) {
        int n = nt * 16 + nloc;
        float cs = csum[n];
        #pragma unroll
        for (int v = 0; v < 8; v++) {
            float o = acc[nt][v] + cs;
            float orl = o > 0.f ? o : 0.f;
            ostage[wave][v + 8 * kh][n] = dorelu ? orl : o;
        }
    }

    // coalesced write-out: 2 lanes per row, 128 B each (8 x b128)
    int rrow = lane >> 1;
    int half = lane & 1;
    int grow = rowBase + rrow;
    const uint4* sp = (const uint4*)&ostage[wave][rrow][half * 32];
    if (grow < NN) {
        uint4* dp = (uint4*)(xnext + (size_t)grow * DD + half * 32);
        #pragma unroll
        for (int i = 0; i < 8; i++) dp[i] = sp[i];
    }
}

// ---------------------------------------------------------------------------
extern "C" void kernel_launch(void* const* d_in, const int* in_sizes, int n_in,
                              void* d_out, int out_size, void* d_ws, size_t ws_size,
                              hipStream_t stream) {
    const float* x    = (const float*)d_in[0];
    const int*   ei   = (const int*)d_in[1];
    const int*   et   = (const int*)d_in[2];
    const float* W1   = (const float*)d_in[3];
    const float* b1   = (const float*)d_in[4];
    const float* g1   = (const float*)d_in[5];
    const float* be1  = (const float*)d_in[6];
    const float* W2   = (const float*)d_in[7];
    const float* b2   = (const float*)d_in[8];
    const float* rW1  = (const float*)d_in[9];
    const float* rb1  = (const float*)d_in[10];
    const float* rg1  = (const float*)d_in[11];
    const float* rbe1 = (const float*)d_in[12];
    const float* rW2  = (const float*)d_in[13];
    const float* rb2  = (const float*)d_in[14];
    const float* bias = (const float*)d_in[15];
    float* out = (float*)d_out;

    // workspace layout (all offsets 32B aligned)
    char* ws = (char*)d_ws;
    size_t off = 0;
    float* h = (float*)(ws + off);          off += (size_t)RR * NN * DD * sizeof(float);          // 204.8 MB
    _Float16* z = (_Float16*)(ws + off);    off += (size_t)(RR + 1) * NN * DD * sizeof(_Float16); // 115.2 MB
    float* xbuf = (float*)(ws + off);       off += (size_t)NN * DD * sizeof(float);               // 25.6 MB
    _Float16* W1t  = (_Float16*)(ws + off); off += (size_t)LL * RR * 4096 * sizeof(_Float16);
    _Float16* W2t  = (_Float16*)(ws + off); off += (size_t)LL * RR * 4096 * sizeof(_Float16);
    _Float16* rW1t = (_Float16*)(ws + off); off += (size_t)LL * 4096 * sizeof(_Float16);
    _Float16* rW2t = (_Float16*)(ws + off); off += (size_t)LL * 4096 * sizeof(_Float16);
    float* ssum  = (float*)(ws + off);      off += (RR + 1) * DD * sizeof(float);
    float* ssq   = (float*)(ws + off);      off += (RR + 1) * DD * sizeof(float);
    float* scl   = (float*)(ws + off);      off += (RR + 1) * DD * sizeof(float);
    float* shf   = (float*)(ws + off);      off += (RR + 1) * DD * sizeof(float);
    float* csum  = (float*)(ws + off);      off += DD * sizeof(float);

    k_weights<<<36, 256, 0, stream>>>(W1, W2, rW1, rW2, W1t, W2t, rW1t, rW2t);

    const int nodeBlocks = (NN + 127) / 128;
    const float* xc = x;
    for (int layer = 0; layer < LL; layer++) {
        hipMemsetAsync(h, 0, (size_t)RR * NN * DD * sizeof(float), stream);
        hipMemsetAsync(ssum, 0, 2 * (RR + 1) * DD * sizeof(float), stream); // ssum+ssq contiguous

        long long stot = (long long)EE * 16;
        k_scatter<<<(unsigned)((stot + 255) / 256), 256, 0, stream>>>(xc, ei, et, h);

        dim3 g2(nodeBlocks, RR + 1);
        k_gemm1<<<g2, 256, 0, stream>>>(h, xc, W1t, rW1t, b1, rb1, z, ssum, ssq, layer);

        k_stats<<<1, 576, 0, stream>>>(ssum, ssq, g1, be1, rg1, rbe1, b2, rb2, bias,
                                       scl, shf, csum, layer);

        float* xn = (layer + 1 < LL) ? xbuf : out;
        k_gemm2<<<nodeBlocks, 256, 0, stream>>>(z, W2t, rW2t, scl, shf, csum, xn,
                                                layer, (layer + 1 < LL) ? 1 : 0);
        xc = xn;
    }
}
